// AFNB_6459630813541
// MI455X (gfx1250) — matrix operand order repaired and checked by reading, not compiled, not run on hardware
//
#include <hip/hip_runtime.h>
#include <hip/hip_bf16.h>
#include <math.h>

typedef __attribute__((ext_vector_type(16))) __bf16 v16bf;
typedef __attribute__((ext_vector_type(8)))  __bf16 v8bf;
typedef __attribute__((ext_vector_type(8)))  float  v8f;

// ---------------------------------------------------------------------------
// Batched GEMM:  Out[b][m][n] = sum_k A[b][m][k] * In[b][k][n]  (+ Res)
//   A:   [M,K] row-major fp32, batch stride aStride (0 = shared weights)
//   In:  [B,K,NN] fp32, Out: [B,M,NN] fp32
// M, K compile-time so all strides fold into immediate offsets; NN = 4096.
// Block = 256 threads (8 waves, wave32). Block tile 64(M) x 128(N), K-step 32.
// LDS tiles are stored in WMMA-fragment order so each lane loads its whole
// 32-byte fragment with one aligned v16bf load (wide ds_load_b128s).
// ---------------------------------------------------------------------------
template<int M, int K, bool RES>
__global__ __launch_bounds__(256)
void afnb_gemm_wmma(const float* __restrict__ A, long long aStride,
                    const float* __restrict__ In, const float* __restrict__ Res,
                    float* __restrict__ Out)
{
    constexpr int NN = 4096;
    // row stride 48 bf16 = 96 B (multiple of 32 B) keeps v16bf loads aligned
    __shared__ __align__(32) __bf16 Abuf[64][48];   // [m][k] layout, contiguous k
    __shared__ __align__(32) __bf16 Bbuf[128][48];  // [n][k] transposed layout

    const int tid    = threadIdx.x;
    const int b      = blockIdx.z;
    const int mBlock = blockIdx.y * 64;
    const int nBlock = blockIdx.x * 128;

    const float* Ab  = A  + (size_t)aStride * b;
    const float* Inb = In + (size_t)b * (K * NN);

    const int wave = tid >> 5;          // 0..7
    const int lane = tid & 31;
    const int wm   = wave & 3;          // M sub-tile (4 x 16 = 64)
    const int wn   = wave >> 2;         // N sub-tile (2 x 64 = 128)
    const int col  = lane & 15;
    const int h    = lane >> 4;         // lane half

    // staging decomposition
    const int ar  = tid & 63;                 // A row (0..63)
    const int akb = (tid >> 6) << 3;          // A k-base: 0,8,16,24
    const int apb = akb;                      // contiguous k base
    const int bc  = tid & 127;                // B column (0..127)
    const int bkb = (tid >> 7) << 4;          // B k-base: 0,16

    const float* aSrc = &Ab[(size_t)(mBlock + ar) * K + akb];
    const float* bSrc = &Inb[(size_t)bkb * NN + nBlock + bc];

    v8f acc[4] = {};

    for (int k0 = 0; k0 < K; k0 += 32) {
        // ---- stage A: 8 contiguous floats -> one 16B LDS store ----
        {
            v8bf t;
            #pragma unroll
            for (int j = 0; j < 8; ++j) t[j] = (__bf16)aSrc[k0 + j];
            *(v8bf*)&Abuf[ar][apb] = t;
        }
        // ---- stage B: 16 k-strided floats (immediate offsets) ----
        {
            const float* src = bSrc + (size_t)k0 * NN;
            v8bf t0, t1;
            #pragma unroll
            for (int j = 0; j < 8; ++j) t0[j] = (__bf16)src[j * NN];
            #pragma unroll
            for (int j = 0; j < 8; ++j) t1[j] = (__bf16)src[(8 + j) * NN];
            *(v8bf*)&Bbuf[bc][bkb]     = t0;
            *(v8bf*)&Bbuf[bc][bkb + 8] = t1;
        }
        // hint: prefetch next K-tile of In (global_prefetch_b8)
        if (k0 + 32 < K) {
            __builtin_prefetch(bSrc + (size_t)(k0 + 32) * NN, 0, 1);
        }
        __syncthreads();

        // ---- fragments: one aligned 32B LDS load each ----
        const int arow = wm * 16 + (lane & 15);
        const v16bf afrag = *(const v16bf*)&Abuf[arow][h * 16];

        #pragma unroll
        for (int t = 0; t < 4; ++t) {
            const int bcol = wn * 64 + t * 16 + col;
            const v16bf bfrag = *(const v16bf*)&Bbuf[bcol][h * 16];
            acc[t] = __builtin_amdgcn_wmma_f32_16x16x32_bf16(
                false, afrag, false, bfrag, (short)0, acc[t], false, false);
        }
        __syncthreads();
    }

    // ---- epilogue: C/D layout -> global (optional residual add) ----
    const size_t oBase = (size_t)b * (M * NN)
                       + (size_t)(mBlock + wm * 16 + h * 8) * NN
                       + nBlock + wn * 64 + col;
    #pragma unroll
    for (int t = 0; t < 4; ++t) {
        #pragma unroll
        for (int r = 0; r < 8; ++r) {
            size_t oidx = oBase + r * NN + t * 16;
            float v = acc[t][r];
            if (RES) v += Res[oidx];
            Out[oidx] = v;
        }
    }
}

// ---------------------------------------------------------------------------
// SPP adaptive max pool: in [B,IC,64,64] -> zero-padded (S=110 -> 128):
//   mode 0: out[b][s][c]  (A-matrix for logits GEMM)
//   mode 1: out[b][c][s]  (A-matrix for agg GEMM)
// One block per (b,c) plane; plane cached in LDS.
// ---------------------------------------------------------------------------
__global__ __launch_bounds__(256)
void afnb_spp_pool(const float* __restrict__ in, float* __restrict__ out, int mode)
{
    __shared__ float plane[4096];
    const int tid = threadIdx.x;
    const int IC = 256, Sp = 128, S = 110;
    const int b = blockIdx.x / IC;
    const int c = blockIdx.x % IC;
    const float* src = in + (size_t)blockIdx.x * 4096;

    #pragma unroll
    for (int i = 0; i < 16; ++i)
        plane[tid + i * 256] = src[tid + i * 256];
    __syncthreads();

    for (int s = tid; s < Sp; s += 256) {
        float val = 0.0f;
        if (s < S) {
            int o, idx;
            if (s == 0)      { o = 1; idx = 0;      }
            else if (s < 10) { o = 3; idx = s - 1;  }
            else if (s < 46) { o = 6; idx = s - 10; }
            else             { o = 8; idx = s - 46; }
            int oy = idx / o, ox = idx % o;
            int h0 = (oy * 64) / o, h1 = ((oy + 1) * 64 + o - 1) / o;
            int w0 = (ox * 64) / o, w1 = ((ox + 1) * 64 + o - 1) / o;
            float mx = -INFINITY;
            for (int hh = h0; hh < h1; ++hh)
                for (int ww = w0; ww < w1; ++ww)
                    mx = fmaxf(mx, plane[hh * 64 + ww]);
            val = mx;
        }
        size_t oidx = (mode == 0)
            ? ((size_t)b * Sp + s) * IC + c
            : ((size_t)b * IC + c) * Sp + s;
        out[oidx] = val;
    }
}

// ---------------------------------------------------------------------------
// Softmax over the QUERY axis: logits[b][s][0..4095] -> P = exp(l-m)/d,
// in place. Pad rows (s >= 110) are zeroed so the K=128 agg GEMM is exact.
// One block per (b,s) row.
// ---------------------------------------------------------------------------
__global__ __launch_bounds__(256)
void afnb_softmax_rows(float* __restrict__ logits)
{
    __shared__ float red[256];
    const int tid = threadIdx.x;
    const int s = blockIdx.x & 127;
    float* row = logits + (size_t)blockIdx.x * 4096;

    if (s >= 110) {
        #pragma unroll
        for (int i = 0; i < 16; ++i) row[tid + i * 256] = 0.0f;
        return;
    }

    float lm = -INFINITY;
    #pragma unroll
    for (int i = 0; i < 16; ++i) lm = fmaxf(lm, row[tid + i * 256]);
    red[tid] = lm;
    __syncthreads();
    for (int off = 128; off > 0; off >>= 1) {
        if (tid < off) red[tid] = fmaxf(red[tid], red[tid + off]);
        __syncthreads();
    }
    const float m = red[0];
    __syncthreads();

    float ls = 0.0f;
    #pragma unroll
    for (int i = 0; i < 16; ++i) ls += __expf(row[tid + i * 256] - m);
    red[tid] = ls;
    __syncthreads();
    for (int off = 128; off > 0; off >>= 1) {
        if (tid < off) red[tid] += red[tid + off];
        __syncthreads();
    }
    const float rinv = 1.0f / red[0];

    #pragma unroll
    for (int i = 0; i < 16; ++i) {
        int j = tid + i * 256;
        row[j] = __expf(row[j] - m) * rinv;
    }
}

// ---------------------------------------------------------------------------
extern "C" void kernel_launch(void* const* d_in, const int* in_sizes, int n_in,
                              void* d_out, int out_size, void* d_ws, size_t ws_size,
                              hipStream_t stream)
{
    const float* x       = (const float*)d_in[0];
    const float* y       = (const float*)d_in[1];
    const float* w_phi   = (const float*)d_in[2];
    const float* w_theta = (const float*)d_in[3];
    const float* w_g     = (const float*)d_in[4];
    const float* w_mask  = (const float*)d_in[5];
    float* out = (float*)d_out;

    const int B = 16, C = 512, IC = 256, Nn = 4096, Sp = 128;
    const size_t szBig = (size_t)B * IC * Nn;   // 16.78M floats

    float* ws      = (float*)d_ws;
    float* buf_phi = ws;                        // x_phi      [B,IC,N]
    float* buf_t   = buf_phi + szBig;           // theta conv -> logits/P [B,Sp,N]
    float* buf_g   = buf_t + szBig;             // g conv     -> agg      [B,IC,N]
    float* ktA     = buf_g + szBig;             // pooled theta [B,Sp,IC]
    float* kgT     = ktA + (size_t)B * Sp * IC; // pooled g     [B,IC,Sp]

    dim3 blk(256);

    // 1–2: theta / g 1x1 convs (shared weights, aStride=0)
    afnb_gemm_wmma<256, 512, false><<<dim3(Nn / 128, IC / 64, B), blk, 0, stream>>>(
        w_theta, 0, y, nullptr, buf_t);
    afnb_gemm_wmma<256, 512, false><<<dim3(Nn / 128, IC / 64, B), blk, 0, stream>>>(
        w_g, 0, y, nullptr, buf_g);

    // 3–4: SPP pooling into zero-padded key/value matrices
    afnb_spp_pool<<<dim3(B * IC), blk, 0, stream>>>(buf_t, ktA, 0);
    afnb_spp_pool<<<dim3(B * IC), blk, 0, stream>>>(buf_g, kgT, 1);

    // 5: phi 1x1 conv
    afnb_gemm_wmma<256, 512, false><<<dim3(Nn / 128, IC / 64, B), blk, 0, stream>>>(
        w_phi, 0, x, nullptr, buf_phi);

    // 6: logits[b][s][n] = sum_c ktA[b][s][c] * x_phi[b][c][n]
    afnb_gemm_wmma<128, 256, false><<<dim3(Nn / 128, Sp / 64, B), blk, 0, stream>>>(
        ktA, (long long)Sp * IC, buf_phi, nullptr, buf_t);

    // 7: softmax over query axis (in place), zero pad rows
    afnb_softmax_rows<<<dim3(B * Sp), blk, 0, stream>>>(buf_t);

    // 8: agg[b][c][n] = sum_s kgT[b][c][s] * P[b][s][n]
    afnb_gemm_wmma<256, 128, false><<<dim3(Nn / 128, IC / 64, B), blk, 0, stream>>>(
        kgT, (long long)IC * Sp, buf_t, nullptr, buf_g);

    // 9: out = w_mask @ agg + x  (residual fused in epilogue)
    afnb_gemm_wmma<512, 256, true><<<dim3(Nn / 128, C / 64, B), blk, 0, stream>>>(
        w_mask, 0, buf_g, x, out);
}
